// MaskedMultiHeadSelfAttention_3152505995624
// MI455X (gfx1250) — matrix operand order repaired and checked by reading, not compiled, not run on hardware
//
#include <hip/hip_runtime.h>
#include <hip/hip_bf16.h>

#define D_MODEL 512
#define NHEADS  8
#define DHEAD   64
#define BATCH   4
#define SEQ     2048
#define MTOT    (BATCH * SEQ)   // 8192

typedef __attribute__((ext_vector_type(16))) __bf16 bf16x16;
typedef __attribute__((ext_vector_type(8)))  float  f32x8;

union FragU { uint4 u[2]; bf16x16 v; };

// ---- CDNA5 async global->LDS copy (ASYNCcnt), guarded with sync fallback ----
#if defined(__has_builtin)
#  if __has_builtin(__builtin_amdgcn_global_load_async_to_lds_b128)
#    define HAVE_ASYNC_LDS 1
#  endif
#  if __has_builtin(__builtin_amdgcn_s_wait_asynccnt)
#    define HAVE_WAIT_ASYNC 1
#  endif
#endif

typedef int v4i __attribute__((vector_size(16)));
typedef __attribute__((address_space(1))) v4i* gv4i_p;
typedef __attribute__((address_space(3))) v4i* lv4i_p;

__device__ __forceinline__ void cp16(void* lds, const void* g) {
#ifdef HAVE_ASYNC_LDS
  __builtin_amdgcn_global_load_async_to_lds_b128(
      (gv4i_p)const_cast<void*>(g), (lv4i_p)lds, 0, 0);
#else
  *(uint4*)lds = *(const uint4*)g;
#endif
}

__device__ __forceinline__ void wait_cp() {
#ifdef HAVE_ASYNC_LDS
#  ifdef HAVE_WAIT_ASYNC
  __builtin_amdgcn_s_wait_asynccnt(0);
#  else
  asm volatile("s_wait_asynccnt 0x0" ::: "memory");
#  endif
#endif
}

__global__ __launch_bounds__(256) void f32_to_bf16_kernel(
    const float* __restrict__ s, __bf16* __restrict__ d, int n) {
  int i = blockIdx.x * blockDim.x + threadIdx.x;
  if (i < n) d[i] = (__bf16)s[i];
}

// -----------------------------------------------------------------------------
// QKV projection: out_bf16[b,h,n,dh] = (h @ W^T + bias) * scale
// WG 256 thr (8 waves). Tile 128(M) x 64(N). K step 32. Double-buffered LDS.
// -----------------------------------------------------------------------------
__global__ __launch_bounds__(256) void qkv_gemm_kernel(
    const __bf16* __restrict__ A, const __bf16* __restrict__ W,
    const float* __restrict__ bias, __bf16* __restrict__ out, float scale) {
  __shared__ __bf16 As[2][128][32];
  __shared__ __bf16 Bs[2][64][32];
  const int tid  = threadIdx.x;
  const int lane = tid & 31;
  const int wave = tid >> 5;
  const int m0   = blockIdx.x * 128;
  const int o0   = blockIdx.y * 64;
  const int mA   = lane & 15;
  const int kb   = (lane >> 4) * 8;   // A-frag k base
  const int bi   = (lane >> 4);       // B-frag uint4 base

  f32x8 acc[4];
#pragma unroll
  for (int i = 0; i < 4; ++i)
#pragma unroll
    for (int e = 0; e < 8; ++e) acc[i][e] = 0.f;

  auto stage = [&](int buf, int k0) {
#pragma unroll
    for (int i = 0; i < 2; ++i) {
      int u = tid * 2 + i;
      int row = u >> 2, cc = u & 3;
      cp16(&As[buf][row][cc * 8], A + (size_t)(m0 + row) * D_MODEL + k0 + cc * 8);
    }
    int row = tid >> 2, cc = tid & 3;
    cp16(&Bs[buf][row][cc * 8], W + (size_t)(o0 + row) * D_MODEL + k0 + cc * 8);
  };

  stage(0, 0);
  const int NK = D_MODEL / 32;
  for (int ki = 0; ki < NK; ++ki) {
    const int buf = ki & 1;
    wait_cp();
    __syncthreads();
    if (ki + 1 < NK) stage(buf ^ 1, (ki + 1) * 32);

    // hoist all fragment loads, then issue WMMAs back-to-back
    FragU af;
    const uint4* arow = (const uint4*)&As[buf][wave * 16 + mA][0];
    af.u[0] = arow[kb >> 3];
    af.u[1] = arow[(kb >> 3) + 2];
    FragU bfr[4];
#pragma unroll
    for (int ct = 0; ct < 4; ++ct) {
      const uint4* brow = (const uint4*)&Bs[buf][ct * 16 + mA][0];
      bfr[ct].u[0] = brow[bi];
      bfr[ct].u[1] = brow[bi + 2];
    }
#pragma unroll
    for (int ct = 0; ct < 4; ++ct)
      acc[ct] = __builtin_amdgcn_wmma_f32_16x16x32_bf16(
          false, af.v, false, bfr[ct].v, (short)0, acc[ct], false, false);
  }

  const int rbase = (lane >> 4) * 8;
#pragma unroll
  for (int ct = 0; ct < 4; ++ct) {
    int o  = o0 + ct * 16 + mA;
    float bsv = bias[o];
    int hh = o >> 6, d = o & 63;
#pragma unroll
    for (int r = 0; r < 8; ++r) {
      int m = m0 + wave * 16 + rbase + r;
      int b = m >> 11, n = m & (SEQ - 1);
      float val = (acc[ct][r] + bsv) * scale;
      out[(((size_t)b * NHEADS + hh) * SEQ + n) * DHEAD + d] = (__bf16)val;
    }
  }
}

// -----------------------------------------------------------------------------
// Flash attention. Grid (N/128, B*H). 8 waves; wave owns 16 Q rows.
// K and V tiles async-staged double-buffered; V transposed LDS->LDS.
// -----------------------------------------------------------------------------
__global__ __launch_bounds__(256) void flash_attn_kernel(
    const __bf16* __restrict__ Q, const __bf16* __restrict__ K,
    const __bf16* __restrict__ V, const int* __restrict__ adj,
    __bf16* __restrict__ ctx) {
  __shared__ __bf16 Ks[2][64][64];   // K tile [kvrow][d]   (16KB)
  __shared__ __bf16 Vn[2][64][64];   // V tile natural      (16KB)
  __shared__ __bf16 Vt[64][64];      // V transposed [d][kv] (8KB)
  __shared__ __bf16 Pb[8][16][64];   // per-wave P          (16KB)

  const int tid  = threadIdx.x;
  const int lane = tid & 31;
  const int wave = tid >> 5;
  const int qblk = blockIdx.x;
  const int bh   = blockIdx.y;
  const size_t base = (size_t)bh * SEQ * DHEAD;
  const int qw0  = qblk * 128 + wave * 16;
  const int mA   = lane & 15;
  const int kbA  = (lane >> 4) * 8;
  const int biB  = (lane >> 4);
  const int rbase = (lane >> 4) * 8;

  // Q fragments (d chunks 0..31 / 32..63), loaded once
  FragU qf[2];
  {
    const uint4* qrow = (const uint4*)(Q + base + (size_t)(qw0 + mA) * DHEAD);
    qf[0].u[0] = qrow[kbA >> 3];
    qf[0].u[1] = qrow[(kbA >> 3) + 2];
    qf[1].u[0] = qrow[4 + (kbA >> 3)];
    qf[1].u[1] = qrow[4 + (kbA >> 3) + 2];
  }

  f32x8 O[4];
#pragma unroll
  for (int i = 0; i < 4; ++i)
#pragma unroll
    for (int e = 0; e < 8; ++e) O[i][e] = 0.f;
  float mrow[8], lrow[8];
#pragma unroll
  for (int r = 0; r < 8; ++r) { mrow[r] = -__builtin_inff(); lrow[r] = 0.f; }

  auto stage = [&](int buf, int k0) {
#pragma unroll
    for (int i = 0; i < 2; ++i) {
      int u = tid * 2 + i;
      int row = u >> 3, cc = u & 7;
      const size_t goff = base + (size_t)(k0 + row) * DHEAD + cc * 8;
      cp16(&Ks[buf][row][cc * 8], K + goff);
      cp16(&Vn[buf][row][cc * 8], V + goff);
    }
  };

  stage(0, 0);
  const int NT = SEQ / 64;
  for (int kt = 0; kt < NT; ++kt) {
    const int buf = kt & 1;
    const int k0  = kt * 64;
    wait_cp();
    __syncthreads();
    if (kt + 1 < NT) stage(buf ^ 1, k0 + 64);

    // transpose V tile: Vn[buf][kv][d] -> Vt[d][kv]
#pragma unroll
    for (int i = 0; i < 2; ++i) {
      int u = tid * 2 + i;
      int vrow = u >> 3, dc = (u & 7) * 8;
      uint4 tmp = *(const uint4*)&Vn[buf][vrow][dc];
      const __bf16* tp = (const __bf16*)&tmp;
#pragma unroll
      for (int j = 0; j < 8; ++j) Vt[dc + j][vrow] = tp[j];
    }
    __syncthreads();

    // S = Q * K^T (4 col tiles; d=64 -> 2 WMMA each); frag loads hoisted
    f32x8 S[4];
#pragma unroll
    for (int ct = 0; ct < 4; ++ct) {
#pragma unroll
      for (int e = 0; e < 8; ++e) S[ct][e] = 0.f;
      FragU kf0, kf1;
      const uint4* krow = (const uint4*)&Ks[buf][ct * 16 + mA][0];
      kf0.u[0] = krow[biB];
      kf0.u[1] = krow[biB + 2];
      kf1.u[0] = krow[4 + biB];
      kf1.u[1] = krow[4 + biB + 2];
      S[ct] = __builtin_amdgcn_wmma_f32_16x16x32_bf16(
          false, qf[0].v, false, kf0.v, (short)0, S[ct], false, false);
      S[ct] = __builtin_amdgcn_wmma_f32_16x16x32_bf16(
          false, qf[1].v, false, kf1.v, (short)0, S[ct], false, false);
    }

    // adjacency mask (Q pre-scaled by 1/sqrt(d))
#pragma unroll
    for (int ct = 0; ct < 4; ++ct) {
      int colg = k0 + ct * 16 + mA;
#pragma unroll
      for (int r = 0; r < 8; ++r) {
        int rowg = qw0 + rbase + r;
        float s = S[ct][r];
        if (adj[(size_t)rowg * SEQ + colg] == 0) s = -__builtin_inff();
        S[ct][r] = s;
      }
    }

    // online softmax stats (row spans 16 lanes of a half-wave)
    float newm[8], alpha[8], rsum[8];
#pragma unroll
    for (int r = 0; r < 8; ++r) {
      float mx = fmaxf(fmaxf(S[0][r], S[1][r]), fmaxf(S[2][r], S[3][r]));
      mx = fmaxf(mx, __shfl_xor(mx, 1, 32));
      mx = fmaxf(mx, __shfl_xor(mx, 2, 32));
      mx = fmaxf(mx, __shfl_xor(mx, 4, 32));
      mx = fmaxf(mx, __shfl_xor(mx, 8, 32));
      float nm = fmaxf(mrow[r], mx);
      alpha[r] = (nm == -__builtin_inff()) ? 1.f : __expf(mrow[r] - nm);
      newm[r] = nm;
      mrow[r] = nm;
      rsum[r] = 0.f;
    }

    // P = exp(S - newm), stash bf16 P in per-wave LDS for A-frag re-swizzle
#pragma unroll
    for (int ct = 0; ct < 4; ++ct) {
#pragma unroll
      for (int r = 0; r < 8; ++r) {
        float s = S[ct][r];
        float p = (s > -1e37f) ? __expf(s - newm[r]) : 0.f;
        rsum[r] += p;
        Pb[wave][rbase + r][ct * 16 + mA] = (__bf16)p;
      }
    }
#pragma unroll
    for (int r = 0; r < 8; ++r) {
      float t = rsum[r];
      t += __shfl_xor(t, 1, 32);
      t += __shfl_xor(t, 2, 32);
      t += __shfl_xor(t, 4, 32);
      t += __shfl_xor(t, 8, 32);
      lrow[r] = lrow[r] * alpha[r] + t;
    }
#pragma unroll
    for (int dt = 0; dt < 4; ++dt)
#pragma unroll
      for (int r = 0; r < 8; ++r) O[dt][r] *= alpha[r];

    // P A-fragments (same wave wrote them; DScnt ordering applies)
    FragU pf[2];
#pragma unroll
    for (int c = 0; c < 2; ++c) {
      const uint4* prow = (const uint4*)&Pb[wave][mA][c * 32];
      pf[c].u[0] = prow[kbA >> 3];
      pf[c].u[1] = prow[(kbA >> 3) + 2];
    }

    // O += P * V (4 d tiles; kv=64 -> 2 WMMA each); frag loads hoisted
#pragma unroll
    for (int dt = 0; dt < 4; ++dt) {
      FragU vf0, vf1;
      const uint4* vrow = (const uint4*)&Vt[dt * 16 + mA][0];
      vf0.u[0] = vrow[biB];
      vf0.u[1] = vrow[biB + 2];
      vf1.u[0] = vrow[4 + biB];
      vf1.u[1] = vrow[4 + biB + 2];
      O[dt] = __builtin_amdgcn_wmma_f32_16x16x32_bf16(
          false, pf[0].v, false, vf0.v, (short)0, O[dt], false, false);
      O[dt] = __builtin_amdgcn_wmma_f32_16x16x32_bf16(
          false, pf[1].v, false, vf1.v, (short)0, O[dt], false, false);
    }
  }

  // normalize and store context (bf16, head layout)
#pragma unroll
  for (int r = 0; r < 8; ++r) {
    float inv = (lrow[r] > 0.f) ? 1.f / lrow[r] : 0.f;
    int row = qw0 + rbase + r;
#pragma unroll
    for (int dt = 0; dt < 4; ++dt) {
      float val = O[dt][r] * inv;
      ctx[base + (size_t)row * DHEAD + dt * 16 + mA] = (__bf16)val;
    }
  }
}

// -----------------------------------------------------------------------------
// Output projection: out_f32[b,n,:] = ctx(head layout) @ Wo^T + bo
// -----------------------------------------------------------------------------
__global__ __launch_bounds__(256) void oproj_gemm_kernel(
    const __bf16* __restrict__ ctx, const __bf16* __restrict__ W,
    const float* __restrict__ bias, float* __restrict__ out) {
  __shared__ __bf16 As[2][128][32];
  __shared__ __bf16 Bs[2][64][32];
  const int tid  = threadIdx.x;
  const int lane = tid & 31;
  const int wave = tid >> 5;
  const int m0   = blockIdx.x * 128;
  const int o0   = blockIdx.y * 64;
  const int mA   = lane & 15;
  const int kb   = (lane >> 4) * 8;
  const int bi   = (lane >> 4);

  f32x8 acc[4];
#pragma unroll
  for (int i = 0; i < 4; ++i)
#pragma unroll
    for (int e = 0; e < 8; ++e) acc[i][e] = 0.f;

  auto stage = [&](int bufi, int k0) {
#pragma unroll
    for (int i = 0; i < 2; ++i) {
      int u = tid * 2 + i;
      int row = u >> 2, cc = u & 3;
      int m = m0 + row, k = k0 + cc * 8;
      int b = m >> 11, n = m & (SEQ - 1);
      cp16(&As[bufi][row][cc * 8],
           ctx + ((((size_t)b * NHEADS + (k >> 6)) * SEQ + n) * DHEAD + (k & 63)));
    }
    int row = tid >> 2, cc = tid & 3;
    cp16(&Bs[bufi][row][cc * 8], W + (size_t)(o0 + row) * D_MODEL + k0 + cc * 8);
  };

  stage(0, 0);
  const int NK = D_MODEL / 32;
  for (int ki = 0; ki < NK; ++ki) {
    const int buf = ki & 1;
    wait_cp();
    __syncthreads();
    if (ki + 1 < NK) stage(buf ^ 1, (ki + 1) * 32);

    FragU af;
    const uint4* arow = (const uint4*)&As[buf][wave * 16 + mA][0];
    af.u[0] = arow[kb >> 3];
    af.u[1] = arow[(kb >> 3) + 2];
    FragU bfr[4];
#pragma unroll
    for (int ct = 0; ct < 4; ++ct) {
      const uint4* brow = (const uint4*)&Bs[buf][ct * 16 + mA][0];
      bfr[ct].u[0] = brow[bi];
      bfr[ct].u[1] = brow[bi + 2];
    }
#pragma unroll
    for (int ct = 0; ct < 4; ++ct)
      acc[ct] = __builtin_amdgcn_wmma_f32_16x16x32_bf16(
          false, af.v, false, bfr[ct].v, (short)0, acc[ct], false, false);
  }

  const int rbase = (lane >> 4) * 8;
#pragma unroll
  for (int ct = 0; ct < 4; ++ct) {
    int o = o0 + ct * 16 + mA;
    float bsv = bias[o];
#pragma unroll
    for (int r = 0; r < 8; ++r) {
      int m = m0 + wave * 16 + rbase + r;
      out[(size_t)m * D_MODEL + o] = acc[ct][r] + bsv;
    }
  }
}

extern "C" void kernel_launch(void* const* d_in, const int* in_sizes, int n_in,
                              void* d_out, int out_size, void* d_ws, size_t ws_size,
                              hipStream_t stream) {
  (void)in_sizes; (void)n_in; (void)out_size; (void)ws_size;
  const float* h   = (const float*)d_in[0];
  const int*   adj = (const int*)d_in[1];
  const float* Wq  = (const float*)d_in[2];
  const float* bq  = (const float*)d_in[3];
  const float* Wk  = (const float*)d_in[4];
  const float* bk  = (const float*)d_in[5];
  const float* Wv  = (const float*)d_in[6];
  const float* bv  = (const float*)d_in[7];
  const float* Wo  = (const float*)d_in[8];
  const float* bo  = (const float*)d_in[9];
  float* out = (float*)d_out;

  char* ws = (char*)d_ws;
  const size_t NH = (size_t)MTOT * D_MODEL;
  const size_t NW = (size_t)D_MODEL * D_MODEL;
  __bf16* hbf = (__bf16*)ws;                     // 8 MB
  __bf16* wqb = (__bf16*)(ws + (8u << 20));
  __bf16* wkb = wqb + NW;
  __bf16* wvb = wkb + NW;
  __bf16* wob = wvb + NW;
  __bf16* Qb  = (__bf16*)(ws + (10u << 20));     // 8 MB each
  __bf16* Kb  = Qb + NH;
  __bf16* Vb  = Kb + NH;
  __bf16* ctx = Vb + NH;

  f32_to_bf16_kernel<<<(int)((NH + 255) / 256), 256, 0, stream>>>(h, hbf, (int)NH);
  f32_to_bf16_kernel<<<(int)((NW + 255) / 256), 256, 0, stream>>>(Wq, wqb, (int)NW);
  f32_to_bf16_kernel<<<(int)((NW + 255) / 256), 256, 0, stream>>>(Wk, wkb, (int)NW);
  f32_to_bf16_kernel<<<(int)((NW + 255) / 256), 256, 0, stream>>>(Wv, wvb, (int)NW);
  f32_to_bf16_kernel<<<(int)((NW + 255) / 256), 256, 0, stream>>>(Wo, wob, (int)NW);

  dim3 ggrid(MTOT / 128, D_MODEL / 64);
  qkv_gemm_kernel<<<ggrid, 256, 0, stream>>>(hbf, wqb, bq, Qb, 0.125f);
  qkv_gemm_kernel<<<ggrid, 256, 0, stream>>>(hbf, wkb, bk, Kb, 1.0f);
  qkv_gemm_kernel<<<ggrid, 256, 0, stream>>>(hbf, wvb, bv, Vb, 1.0f);

  dim3 agrid(SEQ / 128, BATCH * NHEADS);
  flash_attn_kernel<<<agrid, 256, 0, stream>>>(Qb, Kb, Vb, adj, ctx);

  oproj_gemm_kernel<<<ggrid, 256, 0, stream>>>(ctx, wob, bo, out);
}
